// MultiHeadAttention_25202868093354
// MI455X (gfx1250) — matrix operand order repaired and checked by reading, not compiled, not run on hardware
//
#include <hip/hip_runtime.h>

#define SL 2048
#define DM 1024
#define NH 16
#define DKH 64
#define NBS 2

typedef __attribute__((ext_vector_type(16))) __bf16 bf16x16;
typedef __attribute__((ext_vector_type(8)))  float  f32x8;
typedef __attribute__((ext_vector_type(4)))  float  f32x4;
typedef __attribute__((ext_vector_type(4)))  unsigned int u32x4;
typedef __attribute__((ext_vector_type(4)))  int i32x4;
typedef __attribute__((ext_vector_type(8)))  int i32x8;

#if __has_builtin(__builtin_amdgcn_tensor_load_to_lds) && \
    __has_builtin(__builtin_amdgcn_s_wait_tensorcnt)
#define HAS_TDM 1
#else
#define HAS_TDM 0
#endif

union Frag {
  bf16x16 v;
  u32x4   q[2];
};

__device__ __forceinline__ unsigned short f2bf(float x) {
  unsigned int u = __float_as_uint(x);
  u += 0x7fffu + ((u >> 16) & 1u);
  return (unsigned short)(u >> 16);
}

__device__ __forceinline__ f32x8 zero8() {
  f32x8 z;
#pragma unroll
  for (int i = 0; i < 8; ++i) z[i] = 0.0f;
  return z;
}

__device__ __forceinline__ f32x8 wmma_bf16(const Frag& a, const Frag& b, f32x8 c) {
  return __builtin_amdgcn_wmma_f32_16x16x32_bf16(false, a.v, false, b.v,
                                                 (short)0, c, false, false);
}

#if HAS_TDM
// Issue a 2D TDM load (global -> LDS), descriptor per CDNA5 ISA.
// data_size_log2: 0 = 1B elements, 1 = 2B elements.
// pad_interval/pad_amount: ISA-encoded (2^(i+1) DWORDs data, (a+1) DWORDs pad); 0/0 = none.
__device__ __forceinline__ void tdm_load_2d(unsigned lds_addr, const void* gptr,
                                            unsigned data_size_log2,
                                            unsigned tensor_d0, unsigned tensor_d1,
                                            unsigned tile_d0, unsigned tile_d1,
                                            unsigned long long stride0,
                                            unsigned pad_enable, unsigned pad_interval,
                                            unsigned pad_amount) {
  const unsigned long long ga = (unsigned long long)(uintptr_t)gptr;
  u32x4 g0;
  g0[0] = 1u;                                        // count=1 valid user descriptor
  g0[1] = lds_addr;                                  // LDS byte address
  g0[2] = (unsigned)(ga & 0xffffffffu);              // global_addr[31:0]
  g0[3] = (unsigned)((ga >> 32) & 0x1ffffffu) | (2u << 30);  // addr[56:32] | type=2
  i32x8 g1;
  g1[0] = (int)((data_size_log2 << 16) | (pad_enable << 20) |
                (pad_interval << 22) | (pad_amount << 25));
  g1[1] = (int)((tensor_d0 & 0xffffu) << 16);        // [63:48] tensor_dim0 lo16
  g1[2] = (int)(((tensor_d0 >> 16) & 0xffffu) | ((tensor_d1 & 0xffffu) << 16));
  g1[3] = (int)(((tensor_d1 >> 16) & 0xffffu) | ((tile_d0 & 0xffffu) << 16));
  g1[4] = (int)(tile_d1 & 0xffffu);                  // tile_dim1 | tile_dim2=0
  g1[5] = (int)(unsigned)(stride0 & 0xffffffffu);    // tensor_dim0_stride lo32
  g1[6] = (int)(unsigned)((stride0 >> 32) & 0xffffu);// stride hi16 | dim1_stride=0
  g1[7] = 0;
  i32x4 gz;
  gz[0] = 0; gz[1] = 0; gz[2] = 0; gz[3] = 0;
#if __clang_major__ >= 23
  i32x8 gz8;
#pragma unroll
  for (int i = 0; i < 8; ++i) gz8[i] = 0;
  __builtin_amdgcn_tensor_load_to_lds(g0, g1, gz, gz, gz8, 0);
#else
  __builtin_amdgcn_tensor_load_to_lds(g0, g1, gz, gz, 0);
#endif
}
#endif

// ---------------------------------------------------------------------------
// One-shot fp32 -> bf16 conversion (8 elements / thread, exact grids).
// ---------------------------------------------------------------------------
__global__ __launch_bounds__(256) void cvt_bf16(const float* __restrict__ in,
                                                unsigned short* __restrict__ out,
                                                int n) {
  const int i = (blockIdx.x * 256 + threadIdx.x) * 8;
  if (i >= n) return;
  f32x4 a = *(const f32x4*)&in[i];
  f32x4 b = *(const f32x4*)&in[i + 4];
  u32x4 o;
  o[0] = (unsigned)f2bf(a[0]) | ((unsigned)f2bf(a[1]) << 16);
  o[1] = (unsigned)f2bf(a[2]) | ((unsigned)f2bf(a[3]) << 16);
  o[2] = (unsigned)f2bf(b[0]) | ((unsigned)f2bf(b[1]) << 16);
  o[3] = (unsigned)f2bf(b[2]) | ((unsigned)f2bf(b[3]) << 16);
  *(u32x4*)&out[i] = o;
}

// ---------------------------------------------------------------------------
// All-bf16 GEMM: C[M,N] = A[M,K] @ W[N,K]^T + bias  (M = 4096, N = K = 1024)
// A/B tiles (128x32) fetched by the Tensor Data Mover, double-buffered one
// K-step ahead (wave 0 issues, s_wait_tensorcnt syncs). TDM padding gives the
// 40-half LDS row stride (16 DW data + 4 DW pad). Main loop = ds_load + WMMA.
// MODE 0: result bf16 into head layout [b,h,s,d];  MODE 1: result fp32.
// ---------------------------------------------------------------------------
template <int MODE>
__global__ __launch_bounds__(256) void gemm_bb(
    const unsigned short* __restrict__ A, const unsigned short* __restrict__ W,
    const float* __restrict__ bias,
    unsigned short* __restrict__ outb, float* __restrict__ outf) {
  constexpr int LDA = 40;  // halfs: 80B row stride
  __shared__ __align__(16) unsigned short As[2][128 * LDA];  // 2 x 10 KB
  __shared__ __align__(16) unsigned short Bs[2][128 * LDA];  // 2 x 10 KB

  const int tid  = threadIdx.x;
  const int lane = tid & 31;
  const int wave = tid >> 5;
  const int wm = wave >> 2;
  const int wn = wave & 3;
  const int m0 = blockIdx.x * 128;
  const int n0 = blockIdx.y * 128;

  const int l15 = lane & 15;
  const int hiA = (lane >= 16) ? 8 : 0;
  const int hiB = (lane >= 16) ? 16 : 0;
  const int hiR = (lane >= 16) ? 8 : 0;

  f32x8 acc[4][2];
#pragma unroll
  for (int i = 0; i < 4; ++i)
#pragma unroll
    for (int j = 0; j < 2; ++j) acc[i][j] = zero8();

#if HAS_TDM
  if (wave == 0) {
    tdm_load_2d((unsigned)(uintptr_t)&As[0][0], &A[(size_t)m0 * DM],
                1, DM, 4096, 32, 128, DM, 1, /*16 DW*/ 3, /*4 DW*/ 3);
    tdm_load_2d((unsigned)(uintptr_t)&Bs[0][0], &W[(size_t)n0 * DM],
                1, DM, DM, 32, 128, DM, 1, 3, 3);
  }
#endif

  for (int kb = 0; kb < DM; kb += 32) {
    const int p = (kb >> 5) & 1;
#if !HAS_TDM
    {
      const int srow = tid >> 1;
      const int scol = (tid & 1) * 16;
      const u32x4* ga = (const u32x4*)&A[(size_t)(m0 + srow) * DM + kb + scol];
      u32x4* da = (u32x4*)&As[p][srow * LDA + scol];
      da[0] = ga[0]; da[1] = ga[1];
      const u32x4* gb = (const u32x4*)&W[(size_t)(n0 + srow) * DM + kb + scol];
      u32x4* db = (u32x4*)&Bs[p][srow * LDA + scol];
      db[0] = gb[0]; db[1] = gb[1];
    }
#else
    if (wave == 0) {
      if (kb + 32 < DM) {
        tdm_load_2d((unsigned)(uintptr_t)&As[p ^ 1][0], &A[(size_t)m0 * DM + kb + 32],
                    1, DM, 4096, 32, 128, DM, 1, 3, 3);
        tdm_load_2d((unsigned)(uintptr_t)&Bs[p ^ 1][0], &W[(size_t)n0 * DM + kb + 32],
                    1, DM, DM, 32, 128, DM, 1, 3, 3);
        __builtin_amdgcn_s_wait_tensorcnt((short)2);
      } else {
        __builtin_amdgcn_s_wait_tensorcnt((short)0);
      }
    }
#endif
    __syncthreads();

    Frag bfr[2];
#pragma unroll
    for (int nt = 0; nt < 2; ++nt) {
      const u32x4* pb = (const u32x4*)&Bs[p][(wn * 32 + nt * 16 + l15) * LDA + hiA];
      bfr[nt].q[0] = pb[0];
      bfr[nt].q[1] = pb[2];
    }
#pragma unroll
    for (int mt = 0; mt < 4; ++mt) {
      Frag afr;
      const unsigned short* ap = &As[p][(wm * 64 + mt * 16 + l15) * LDA];
      afr.q[0] = *(const u32x4*)&ap[hiA];
      afr.q[1] = *(const u32x4*)&ap[16 + hiA];
#pragma unroll
      for (int nt = 0; nt < 2; ++nt)
        acc[mt][nt] = wmma_bf16(afr, bfr[nt], acc[mt][nt]);
    }
    __syncthreads();  // all reads of buffer p done before its TDM reissue
  }

#pragma unroll
  for (int nt = 0; nt < 2; ++nt) {
    const int n = n0 + wn * 32 + nt * 16 + l15;
    const float bv = bias[n];
#pragma unroll
    for (int mt = 0; mt < 4; ++mt) {
#pragma unroll
      for (int r = 0; r < 8; ++r) {
        const int m = m0 + wm * 64 + mt * 16 + r + hiR;
        const float vv = acc[mt][nt][r] + bv;
        if (MODE == 0) {
          const int b = m >> 11, s = m & (SL - 1);
          const int h = n >> 6, d = n & (DKH - 1);
          outb[((size_t)(b * NH + h) * SL + s) * DKH + d] = f2bf(vv);
        } else {
          outf[(size_t)m * DM + n] = vv;
        }
      }
    }
  }
}

// ---------------------------------------------------------------------------
// Flash-style attention. Block = 256 threads (8 waves), 128 query rows,
// one (b,h) per block, KV streamed in steps of 64.
// K tile + mask tile fetched by the TDM (double-buffered one step ahead).
// ---------------------------------------------------------------------------
__global__ __launch_bounds__(256) void attn_kernel(
    const unsigned short* __restrict__ Qh, const unsigned short* __restrict__ Kh,
    const unsigned short* __restrict__ Vh, const unsigned char* __restrict__ mask,
    unsigned short* __restrict__ Zc) {
  constexpr int LDK = 72;  // 144B row stride (32 DW data + 4 DW TDM pad)
  constexpr int LDV = 72;
  constexpr int LDP = 72;
  __shared__ __align__(16) unsigned short Ks[2][64 * LDK];   // 2 x 9 KB
  __shared__ __align__(16) unsigned char  Ms[2][128 * 64];   // 2 x 8 KB
  __shared__ __align__(16) unsigned short Vt[64 * LDV];      // 9 KB
  __shared__ __align__(16) unsigned short Ps[8 * 16 * LDP];  // 18 KB

  const int tid  = threadIdx.x;
  const int lane = tid & 31;
  const int wave = tid >> 5;
  const int l15 = lane & 15;
  const int hiA = (lane >= 16) ? 8 : 0;
  const int hiB = (lane >= 16) ? 16 : 0;
  const int hiR = (lane >= 16) ? 8 : 0;

  const int qb = blockIdx.x * 128;
  const int bh = blockIdx.y;
  const int b  = bh >> 4;
  const int h  = bh & 15;

  const unsigned short* Qp = Qh + (size_t)bh * SL * DKH;
  const unsigned short* Kp = Kh + (size_t)bh * SL * DKH;
  const unsigned short* Vp = Vh + (size_t)bh * SL * DKH;
  const unsigned char*  Mp = mask + (size_t)b * SL * SL + (size_t)qb * SL;

  const int qrow0 = qb + wave * 16;

  Frag qf[2];
  {
    const int qr = qrow0 + l15;
#pragma unroll
    for (int dc = 0; dc < 2; ++dc) {
      const unsigned short* p = &Qp[(size_t)qr * DKH + dc * 32 + hiA];
      qf[dc].q[0] = *(const u32x4*)&p[0];
      qf[dc].q[1] = *(const u32x4*)&p[16];
    }
  }

  f32x8 zacc[4];
#pragma unroll
  for (int i = 0; i < 4; ++i) zacc[i] = zero8();
  float m_r[8], l_r[8];
#pragma unroll
  for (int r = 0; r < 8; ++r) { m_r[r] = -3.0e38f; l_r[r] = 0.0f; }

#if HAS_TDM
  if (wave == 0) {
    tdm_load_2d((unsigned)(uintptr_t)&Ks[0][0], Kp,
                1, DKH, SL, DKH, 64, DKH, 1, /*32 DW*/ 4, /*4 DW*/ 3);
    tdm_load_2d((unsigned)(uintptr_t)&Ms[0][0], Mp,
                0, SL, SL, 64, 128, SL, 0, 0, 0);
  }
#endif

  for (int kv = 0; kv < SL; kv += 64) {
    const int p = (kv >> 6) & 1;
#if !HAS_TDM
    {
      const int row = tid >> 2;
      const int seg = (tid & 3) * 16;
      const u32x4* g = (const u32x4*)&Kp[(size_t)(kv + row) * DKH + seg];
      u32x4* d = (u32x4*)&Ks[p][row * LDK + seg];
      d[0] = g[0]; d[1] = g[1];
    }
    {
      const int row = tid >> 1;
      const int seg = (tid & 1) * 32;
      const u32x4* g = (const u32x4*)&Mp[(size_t)row * SL + kv + seg];
      u32x4* d = (u32x4*)&Ms[p][row * 64 + seg];
      d[0] = g[0]; d[1] = g[1];
    }
#endif
    // stage V tile transposed: Vt[d][k] (coalesced global reads)
    {
      const int dd = tid & 63;
      const int k0 = (tid >> 6) * 16;
#pragma unroll
      for (int k = 0; k < 16; ++k)
        Vt[dd * LDV + k0 + k] = Vp[(size_t)(kv + k0 + k) * DKH + dd];
    }
#if HAS_TDM
    if (wave == 0) {
      if (kv + 64 < SL) {
        tdm_load_2d((unsigned)(uintptr_t)&Ks[p ^ 1][0], Kp + (size_t)(kv + 64) * DKH,
                    1, DKH, SL, DKH, 64, DKH, 1, 4, 3);
        tdm_load_2d((unsigned)(uintptr_t)&Ms[p ^ 1][0], Mp + (kv + 64),
                    0, SL, SL, 64, 128, SL, 0, 0, 0);
        __builtin_amdgcn_s_wait_tensorcnt((short)2);
      } else {
        __builtin_amdgcn_s_wait_tensorcnt((short)0);
      }
    }
#endif
    __syncthreads();

    // S = Q @ K^T
    f32x8 s[4];
#pragma unroll
    for (int nt = 0; nt < 4; ++nt) s[nt] = zero8();
#pragma unroll
    for (int dc = 0; dc < 2; ++dc) {
#pragma unroll
      for (int nt = 0; nt < 4; ++nt) {
        Frag kf;
        const u32x4* kp = (const u32x4*)&Ks[p][(nt * 16 + l15) * LDK + dc * 32 + hiA];
        kf.q[0] = kp[0];
        kf.q[1] = kp[2];
        s[nt] = wmma_bf16(qf[dc], kf, s[nt]);
      }
    }

    // scale + mask (mask tile in LDS)
#pragma unroll
    for (int nt = 0; nt < 4; ++nt) {
#pragma unroll
      for (int r = 0; r < 8; ++r) {
        const int qloc = wave * 16 + r + hiR;
        const int kcol = nt * 16 + l15;
        float vv = s[nt][r] * 0.125f;  // 1/sqrt(64)
        if (Ms[p][qloc * 64 + kcol]) vv = -1.0e10f;
        s[nt][r] = vv;
      }
    }

    // online softmax: 16-lane xor reductions, row stats per C-layout row
    float mt[8];
#pragma unroll
    for (int r = 0; r < 8; ++r) mt[r] = -3.0e38f;
#pragma unroll
    for (int nt = 0; nt < 4; ++nt)
#pragma unroll
      for (int r = 0; r < 8; ++r) mt[r] = fmaxf(mt[r], s[nt][r]);
#pragma unroll
    for (int msk = 1; msk <= 8; msk <<= 1)
#pragma unroll
      for (int r = 0; r < 8; ++r) mt[r] = fmaxf(mt[r], __shfl_xor(mt[r], msk, 32));

    float scl[8];
#pragma unroll
    for (int r = 0; r < 8; ++r) {
      const float mn = fmaxf(m_r[r], mt[r]);
      scl[r] = __expf(m_r[r] - mn);
      m_r[r] = mn;
    }
    float ls[8];
#pragma unroll
    for (int r = 0; r < 8; ++r) ls[r] = 0.0f;
#pragma unroll
    for (int nt = 0; nt < 4; ++nt)
#pragma unroll
      for (int r = 0; r < 8; ++r) {
        const float pe = __expf(s[nt][r] - m_r[r]);
        s[nt][r] = pe;
        ls[r] += pe;
      }
#pragma unroll
    for (int msk = 1; msk <= 8; msk <<= 1)
#pragma unroll
      for (int r = 0; r < 8; ++r) ls[r] += __shfl_xor(ls[r], msk, 32);
#pragma unroll
    for (int r = 0; r < 8; ++r) l_r[r] = l_r[r] * scl[r] + ls[r];
#pragma unroll
    for (int nt = 0; nt < 4; ++nt)
#pragma unroll
      for (int r = 0; r < 8; ++r) zacc[nt][r] *= scl[r];

    // C-layout -> A-layout via per-wave LDS region
    unsigned short* pw = &Ps[wave * 16 * LDP];
#pragma unroll
    for (int nt = 0; nt < 4; ++nt)
#pragma unroll
      for (int r = 0; r < 8; ++r)
        pw[(r + hiR) * LDP + nt * 16 + l15] = f2bf(s[nt][r]);

    // O += P @ V
#pragma unroll
    for (int kc = 0; kc < 2; ++kc) {
      Frag pf;
      const unsigned short* pr = &pw[l15 * LDP + kc * 32 + hiA];
      pf.q[0] = *(const u32x4*)&pr[0];
      pf.q[1] = *(const u32x4*)&pr[16];
#pragma unroll
      for (int nt = 0; nt < 4; ++nt) {
        Frag vf;
        const u32x4* vp = (const u32x4*)&Vt[(nt * 16 + l15) * LDV + kc * 32 + hiA];
        vf.q[0] = vp[0];
        vf.q[1] = vp[2];
        zacc[nt] = wmma_bf16(pf, vf, zacc[nt]);
      }
    }
    __syncthreads();
  }

  // epilogue: normalize, write Zc[b, s, h*64 + d] as bf16
#pragma unroll
  for (int nt = 0; nt < 4; ++nt) {
#pragma unroll
    for (int r = 0; r < 8; ++r) {
      const int qq = qrow0 + r + hiR;
      const int dd = nt * 16 + l15;
      const float vv = zacc[nt][r] / fmaxf(l_r[r], 1.0e-20f);
      Zc[((size_t)(b * SL + qq)) * DM + h * DKH + dd] = f2bf(vv);
    }
  }
}

// ---------------------------------------------------------------------------
extern "C" void kernel_launch(void* const* d_in, const int* in_sizes, int n_in,
                              void* d_out, int out_size, void* d_ws, size_t ws_size,
                              hipStream_t stream) {
  (void)in_sizes; (void)n_in; (void)out_size; (void)ws_size;
  const float* q  = (const float*)d_in[0];
  const float* k  = (const float*)d_in[1];
  const float* v  = (const float*)d_in[2];
  const unsigned char* mask = (const unsigned char*)d_in[3];  // jax bool -> 1 byte
  const float* Wq = (const float*)d_in[4];
  const float* bq = (const float*)d_in[5];
  const float* Wk = (const float*)d_in[6];
  const float* bk = (const float*)d_in[7];
  const float* Wv = (const float*)d_in[8];
  const float* bv = (const float*)d_in[9];
  const float* Wo = (const float*)d_in[10];
  const float* bo = (const float*)d_in[11];
  float* out = (float*)d_out;

  // workspace (bf16 elements): heads + converted activations + converted weights
  unsigned short* ws = (unsigned short*)d_ws;
  const size_t HEL = (size_t)NBS * SL * DM;  // 4M
  const size_t WEL = (size_t)DM * DM;        // 1M
  unsigned short* Qh  = ws;
  unsigned short* Kh  = ws + HEL;
  unsigned short* Vh  = ws + 2 * HEL;
  unsigned short* Zc  = ws + 3 * HEL;
  unsigned short* qb16 = ws + 4 * HEL;
  unsigned short* kb16 = ws + 5 * HEL;
  unsigned short* vb16 = ws + 6 * HEL;
  unsigned short* Wqb = ws + 7 * HEL;
  unsigned short* Wkb = Wqb + WEL;
  unsigned short* Wvb = Wqb + 2 * WEL;
  unsigned short* Wob = Wqb + 3 * WEL;

  const dim3 blk(256);
  const int gact = (int)(HEL / (256 * 8));  // 2048
  const int gwt  = (int)(WEL / (256 * 8));  // 512
  cvt_bf16<<<gact, blk, 0, stream>>>(q, qb16, (int)HEL);
  cvt_bf16<<<gact, blk, 0, stream>>>(k, kb16, (int)HEL);
  cvt_bf16<<<gact, blk, 0, stream>>>(v, vb16, (int)HEL);
  cvt_bf16<<<gwt,  blk, 0, stream>>>(Wq, Wqb, (int)WEL);
  cvt_bf16<<<gwt,  blk, 0, stream>>>(Wk, Wkb, (int)WEL);
  cvt_bf16<<<gwt,  blk, 0, stream>>>(Wv, Wvb, (int)WEL);
  cvt_bf16<<<gwt,  blk, 0, stream>>>(Wo, Wob, (int)WEL);

  const dim3 gproj(32, 8);
  gemm_bb<0><<<gproj, blk, 0, stream>>>(qb16, Wqb, bq, Qh, nullptr);
  gemm_bb<0><<<gproj, blk, 0, stream>>>(kb16, Wkb, bk, Kh, nullptr);
  gemm_bb<0><<<gproj, blk, 0, stream>>>(vb16, Wvb, bv, Vh, nullptr);

  const dim3 gattn(SL / 128, NBS * NH);
  attn_kernel<<<gattn, blk, 0, stream>>>(Qh, Kh, Vh, mask, Zc);

  gemm_bb<1><<<gproj, blk, 0, stream>>>(nullptr ? nullptr : Zc, Wob, bo, nullptr, out);
}
